// BatchDynamicSoftLabelAssigner_10462540333798
// MI455X (gfx1250) — compile-verified, matrix operand order checked by reading, not run on hardware
//
#include <hip/hip_runtime.h>
#include <hip/hip_bf16.h>
#include <limits.h>

// Problem constants (match reference)
#define B_   8
#define G_   100
#define NC_  80
#define N_   33600        // 160^2 + 80^2 + 40^2, divisible by 16 (2100 tiles)
#define NT_  2100
#define TOPK_ 13
#define INF_  1.0e8f
#define BN_  (B_ * N_)

typedef __attribute__((ext_vector_type(16))) _Float16 v16h;
typedef __attribute__((ext_vector_type(8)))  float    v8f;

// ---------------------------------------------------------------- helpers
__device__ __forceinline__ float iou_pair(float px1, float py1, float px2, float py2,
                                          float gx1, float gy1, float gx2, float gy2) {
  float a1 = (px2 - px1) * (py2 - py1);
  float a2 = (gx2 - gx1) * (gy2 - gy1);
  float iw = fminf(px2, gx2) - fmaxf(px1, gx1);
  float ih = fminf(py2, gy2) - fmaxf(py1, gy1);
  iw = fmaxf(iw, 0.f); ih = fmaxf(ih, 0.f);
  float ov = iw * ih;
  float un = a1 + a2 - ov;
  // fast divide: v_rcp_f32 + mul instead of IEEE div_scale/div_fmas/div_fixup
  return ov * __builtin_amdgcn_rcpf(fmaxf(un, 1e-6f));
}

// cost = bce*scale^2 + iou_cost + soft_center_prior, INF where anchor invalid
__device__ __forceinline__ float cost_fn(float iou, float x, float dscaled, float vm) {
  float scp = __expf((dscaled - 3.0f) * 2.302585093f);        // 10^(d-3)
  float ic  = -__logf(iou + 1e-7f) * 3.0f;                    // IOU_WEIGHT
  float sig = __builtin_amdgcn_rcpf(1.f + __expf(-x));        // sigmoid
  float sp  = fmaxf(x, 0.f) + __logf(1.f + __expf(-fabsf(x))); // softplus (stable, fast)
  float bce = sp - x * iou;
  float sc  = iou - sig;
  float c   = bce * sc * sc + ic + scp;
  return (vm > 0.f) ? c : INF_;
}

// ---------------------------------------------------------------- kernel 0
// per-(b,n): valid_mask = any gt with center strictly inside & gt valid.
// Also zeroes the 4-word match bitmask.
__global__ __launch_bounds__(256) void k0_validmask(
    const float* __restrict__ priors, const float* __restrict__ gtb,
    const float* __restrict__ flag, float* __restrict__ vmask,
    unsigned* __restrict__ mmask) {
  int b = blockIdx.y;
  int n = blockIdx.x * 256 + threadIdx.x;
  __shared__ float g5[G_][5];
  for (int i = threadIdx.x; i < G_; i += 256) {
    const float* p = gtb + ((size_t)b * G_ + i) * 4;
    g5[i][0] = p[0]; g5[i][1] = p[1]; g5[i][2] = p[2]; g5[i][3] = p[3];
    g5[i][4] = flag[b * G_ + i];
  }
  __syncthreads();
  if (n >= N_) return;
  float cx = priors[n * 4 + 0], cy = priors[n * 4 + 1];
  bool any = false;
  for (int g = 0; g < G_; ++g) {
    float l  = cx - g5[g][0], t  = cy - g5[g][1];
    float rr = g5[g][2] - cx, bb = g5[g][3] - cy;
    float mn = fminf(fminf(l, t), fminf(rr, bb));
    any = any || ((mn > 0.f) && (g5[g][4] > 0.f));
  }
  size_t o = (size_t)b * N_ + n;
  vmask[o] = any ? 1.f : 0.f;
  mmask[o * 4 + 0] = 0u; mmask[o * 4 + 1] = 0u;
  mmask[o * 4 + 2] = 0u; mmask[o * 4 + 3] = 0u;
}

// ---------------------------------------------------------------- kernel 1
// One workgroup per (b,g) column: top-13 IoUs -> dyn_k, then the dyn_k
// lowest-cost anchors (argsort tie-break = lower index) -> match bitmask.
__global__ __launch_bounds__(256) void k1_column(
    const float* __restrict__ pred_b, const float* __restrict__ pred_s,
    const float* __restrict__ priors, const int* __restrict__ gt_lab,
    const float* __restrict__ gt_b, const float* __restrict__ flag,
    const float* __restrict__ vmask, unsigned* __restrict__ mmask) {
  int blk = blockIdx.x;
  int b = blk / G_, g = blk % G_;
  if (flag[b * G_ + g] <= 0.f) return;   // matching &= valid_gt

  const float* gp = gt_b + ((size_t)b * G_ + g) * 4;
  float gx1 = gp[0], gy1 = gp[1], gx2 = gp[2], gy2 = gp[3];
  float gcx = (gx1 + gx2) * 0.5f, gcy = (gy1 + gy2) * 0.5f;
  int lbl = gt_lab[b * G_ + g];
  int tid = threadIdx.x;

  float tv[TOPK_]; int tvi[TOPK_];   // top IoUs (desc)
  float tc[TOPK_]; int tci[TOPK_];   // lowest costs (asc)
#pragma unroll
  for (int j = 0; j < TOPK_; ++j) {
    tv[j] = -1e30f; tvi[j] = INT_MAX;
    tc[j] = 3.4e38f; tci[j] = INT_MAX;
  }

  for (int n = tid; n < N_; n += 256) {
    size_t o = (size_t)b * N_ + n;
    if (n + 256 < N_) __builtin_prefetch(pred_b + (o + 256) * 4, 0, 0);
    const float* pb = pred_b + o * 4;
    float iou = iou_pair(pb[0], pb[1], pb[2], pb[3], gx1, gy1, gx2, gy2);
    float cx = priors[n * 4], cy = priors[n * 4 + 1], st = priors[n * 4 + 2];
    float vm = vmask[o];
    float dx = cx - gcx, dy = cy - gcy;
    float d  = __builtin_amdgcn_sqrtf(dx * dx + dy * dy) * __builtin_amdgcn_rcpf(st) * vm;
    float x  = pred_s[o * NC_ + lbl];
    float c  = cost_fn(iou, x, d, vm);
    // insert iou (descending, tie -> lower index)
    {
      float iv = iou; int ii = n;
#pragma unroll
      for (int j = 0; j < TOPK_; ++j) {
        bool sw = (iv > tv[j]) || (iv == tv[j] && ii < tvi[j]);
        float nv = sw ? iv : tv[j]; int ni = sw ? ii : tvi[j];
        iv = sw ? tv[j] : iv; ii = sw ? tvi[j] : ii;
        tv[j] = nv; tvi[j] = ni;
      }
    }
    // insert cost (ascending, tie -> lower index)
    {
      float cv = c; int ci = n;
#pragma unroll
      for (int j = 0; j < TOPK_; ++j) {
        bool sw = (cv < tc[j]) || (cv == tc[j] && ci < tci[j]);
        float nv = sw ? cv : tc[j]; int ni = sw ? ci : tci[j];
        cv = sw ? tc[j] : cv; ci = sw ? tci[j] : ci;
        tc[j] = nv; tci[j] = ni;
      }
    }
  }

  __shared__ float rv[256];
  __shared__ int   ri[256];

  // Phase A: sum of global top-13 IoUs -> dyn_k
  float s = 0.f;
  for (int round = 0; round < TOPK_; ++round) {
    float lv = -3e38f; int li = INT_MAX;
#pragma unroll
    for (int j = 0; j < TOPK_; ++j) {
      bool bt = (tv[j] > lv) || (tv[j] == lv && tvi[j] < li);
      if (bt) { lv = tv[j]; li = tvi[j]; }
    }
    rv[tid] = lv; ri[tid] = li;
    __syncthreads();
    for (int st2 = 128; st2 > 0; st2 >>= 1) {
      if (tid < st2) {
        float ov = rv[tid + st2]; int oi = ri[tid + st2];
        if (ov > rv[tid] || (ov == rv[tid] && oi < ri[tid])) { rv[tid] = ov; ri[tid] = oi; }
      }
      __syncthreads();
    }
    float wv = rv[0]; int wi = ri[0];
    __syncthreads();
    if (wi != INT_MAX && wv > 0.f) s += wv;
#pragma unroll
    for (int j = 0; j < TOPK_; ++j)
      if (tvi[j] == wi && wi != INT_MAX) { tv[j] = -1e30f; tvi[j] = INT_MAX; }
  }
  int dynk = (int)s;
  dynk = dynk < 1 ? 1 : (dynk > TOPK_ ? TOPK_ : dynk);

  // Phase B: dyn_k smallest costs -> scatter match bits
  for (int round = 0; round < TOPK_; ++round) {
    float lv = 3.4e38f; int li = INT_MAX;
#pragma unroll
    for (int j = 0; j < TOPK_; ++j) {
      bool bt = (tc[j] < lv) || (tc[j] == lv && tci[j] < li);
      if (bt) { lv = tc[j]; li = tci[j]; }
    }
    rv[tid] = lv; ri[tid] = li;
    __syncthreads();
    for (int st2 = 128; st2 > 0; st2 >>= 1) {
      if (tid < st2) {
        float ov = rv[tid + st2]; int oi = ri[tid + st2];
        if (ov < rv[tid] || (ov == rv[tid] && oi < ri[tid])) { rv[tid] = ov; ri[tid] = oi; }
      }
      __syncthreads();
    }
    int wi = ri[0];
    __syncthreads();
    if (tid == 0 && round < dynk && wi != INT_MAX)
      atomicOr(&mmask[((size_t)b * N_ + wi) * 4 + (g >> 5)], 1u << (g & 31));
#pragma unroll
    for (int j = 0; j < TOPK_; ++j)
      if (tci[j] == wi && wi != INT_MAX) { tc[j] = 3.4e38f; tci[j] = INT_MAX; }
  }
}

// ---------------------------------------------------------------- kernel 2
// One wave32 per 16-anchor tile. Score rows staged to LDS with CDNA5 async
// global->LDS B128 copies (ASYNCcnt). Logits x[n,g] = S(16x80) @ OneHot(80x16)
// via 3 chained v_wmma_f32_16x16x32_f16 per 16-g tile (K padded 80->96).
// Then elementwise cost, row argmin + matched-bit resolution, final outputs.
__global__ __launch_bounds__(32) void k2_assign(
    const float* __restrict__ pred_b, const float* __restrict__ pred_s,
    const float* __restrict__ priors, const int* __restrict__ gt_lab,
    const float* __restrict__ gt_b, const float* __restrict__ vmask,
    const unsigned* __restrict__ mmask, float* __restrict__ out) {
  int blk = blockIdx.x;
  int b = blk / NT_;
  int n0 = (blk % NT_) * 16;
  int lane = threadIdx.x;
  int hi = lane >> 4, lm = lane & 15;

  __shared__ __align__(16) float scf[16 * NC_];   // 16 anchors x 80 classes (f32)
  __shared__ float gb[G_ * 4];
  __shared__ float gc[G_ * 2];
  __shared__ int   gl[G_];

  // --- async tensor-path staging: 320 dword4 transfers, 10 per lane ---
  {
    // wave-relative LDS byte offset of scf (low 32 bits of the flat pointer)
    unsigned sbase = (unsigned)(unsigned long long)(void*)&scf[0];
    unsigned long long gbase =
        (unsigned long long)(const void*)(pred_s + ((size_t)b * N_ + n0) * NC_);
#pragma unroll
    for (int it = 0; it < 10; ++it) {
      int e = it * 32 + lane;                    // dword4 index within tile
      unsigned ldsa = sbase + (unsigned)e * 16u;
      unsigned long long ga = gbase + (unsigned long long)e * 16ull;
      asm volatile("global_load_async_to_lds_b128 %0, %1, off"
                   :: "v"(ldsa), "v"(ga) : "memory");
    }
  }

  for (int i = lane; i < G_; i += 32) {
    const float* p = gt_b + ((size_t)b * G_ + i) * 4;
    float x1 = p[0], y1 = p[1], x2 = p[2], y2 = p[3];
    gb[i * 4 + 0] = x1; gb[i * 4 + 1] = y1; gb[i * 4 + 2] = x2; gb[i * 4 + 3] = y2;
    gc[i * 2 + 0] = (x1 + x2) * 0.5f; gc[i * 2 + 1] = (y1 + y2) * 0.5f;
    gl[i] = gt_lab[b * G_ + i];
  }
  asm volatile("s_wait_asynccnt 0" ::: "memory");
  __syncthreads();

  // A operand: 16x32 f16 chunks, lane = M=lm, hi selects K sub-bank.
  // Classes >= 80 are zero padding (only chunk 2 can hit them).
  v16h a0 = {}, a1 = {}, a2 = {};
#pragma unroll
  for (int v = 0; v < 8; ++v) {
    int kb = ((v < 4) ? (2 * v) : (2 * v + 8)) + 8 * hi;
    a0[2 * v]     = (_Float16)scf[lm * NC_ + kb];
    a0[2 * v + 1] = (_Float16)scf[lm * NC_ + kb + 1];
    a1[2 * v]     = (_Float16)scf[lm * NC_ + 32 + kb];
    a1[2 * v + 1] = (_Float16)scf[lm * NC_ + 32 + kb + 1];
    int c2 = 64 + kb;
    a2[2 * v]     = (c2     < NC_) ? (_Float16)scf[lm * NC_ + c2]     : (_Float16)0.f;
    a2[2 * v + 1] = (c2 + 1 < NC_) ? (_Float16)scf[lm * NC_ + c2 + 1] : (_Float16)0.f;
  }

  // Per-lane anchor row data: C/D layout row M = r + 8*hi
  float px1[8], py1[8], px2[8], py2[8], acx[8], acy[8], ars[8], avm[8];
  unsigned mw0[8], mw1[8], mw2[8], mw3[8];
#pragma unroll
  for (int r = 0; r < 8; ++r) {
    int n = n0 + r + 8 * hi;
    size_t o = (size_t)b * N_ + n;
    const float* pb = pred_b + o * 4;
    px1[r] = pb[0]; py1[r] = pb[1]; px2[r] = pb[2]; py2[r] = pb[3];
    acx[r] = priors[n * 4]; acy[r] = priors[n * 4 + 1];
    ars[r] = __builtin_amdgcn_rcpf(priors[n * 4 + 2]);   // 1/stride
    avm[r] = vmask[o];
    mw0[r] = mmask[o * 4 + 0]; mw1[r] = mmask[o * 4 + 1];
    mw2[r] = mmask[o * 4 + 2]; mw3[r] = mmask[o * 4 + 3];
  }

  float mnC[8], mnIou[8], smIou[8];
  int mnI[8], mMG[8], cnt[8];
#pragma unroll
  for (int r = 0; r < 8; ++r) {
    mnC[r] = 3.4e38f; mnIou[r] = 0.f; smIou[r] = 0.f;
    mnI[r] = 0x3FFFFFFF; mMG[r] = 0x3FFFFFFF; cnt[r] = 0;
  }

  for (int t = 0; t < 7; ++t) {
    int g = t * 16 + lm;
    bool gok = (g < G_);
    int gi = gok ? g : 0;
    float gx1 = gb[gi * 4], gy1 = gb[gi * 4 + 1], gx2 = gb[gi * 4 + 2], gy2 = gb[gi * 4 + 3];
    float gcx = gc[gi * 2], gcy = gc[gi * 2 + 1];
    int lbl = gok ? gl[gi] : -1;

    // B operand: one-hot(label) 32x16 f16 chunks, built branchlessly in regs
    v16h b0 = {}, b1 = {}, b2 = {};
#pragma unroll
    for (int v = 0; v < 8; ++v) {
      int kb = ((v < 4) ? (2 * v) : (2 * v + 8)) + 8 * hi;
      b0[2 * v]     = (_Float16)(lbl == kb          ? 1.f : 0.f);
      b0[2 * v + 1] = (_Float16)(lbl == kb + 1      ? 1.f : 0.f);
      b1[2 * v]     = (_Float16)(lbl == 32 + kb     ? 1.f : 0.f);
      b1[2 * v + 1] = (_Float16)(lbl == 32 + kb + 1 ? 1.f : 0.f);
      b2[2 * v]     = (_Float16)(lbl == 64 + kb     ? 1.f : 0.f);
      b2[2 * v + 1] = (_Float16)(lbl == 64 + kb + 1 ? 1.f : 0.f);
    }
    v8f acc = {};
    acc = __builtin_amdgcn_wmma_f32_16x16x32_f16(false, a0, false, b0, (short)0, acc, false, false);
    acc = __builtin_amdgcn_wmma_f32_16x16x32_f16(false, a1, false, b1, (short)0, acc, false, false);
    acc = __builtin_amdgcn_wmma_f32_16x16x32_f16(false, a2, false, b2, (short)0, acc, false, false);

#pragma unroll
    for (int r = 0; r < 8; ++r) {
      float x = acc[r];
      float iou = iou_pair(px1[r], py1[r], px2[r], py2[r], gx1, gy1, gx2, gy2);
      float dx = acx[r] - gcx, dy = acy[r] - gcy;
      float d = __builtin_amdgcn_sqrtf(dx * dx + dy * dy) * ars[r] * avm[r];
      float c = gok ? cost_fn(iou, x, d, avm[r]) : 3.4e38f;
      bool bet = gok && ((c < mnC[r]) || (c == mnC[r] && g < mnI[r]));
      if (bet) { mnC[r] = c; mnI[r] = g; mnIou[r] = iou; }
      int wsl = g >> 5;
      unsigned w = (wsl == 0) ? mw0[r] : ((wsl == 1) ? mw1[r] : ((wsl == 2) ? mw2[r] : mw3[r]));
      unsigned bit = gok ? ((w >> (g & 31)) & 1u) : 0u;
      if (bit) { cnt[r] += 1; smIou[r] += iou; mMG[r] = (g < mMG[r]) ? g : mMG[r]; }
    }
  }

  // Butterfly reductions within each 16-lane half
#pragma unroll
  for (int r = 0; r < 8; ++r) {
    for (int off = 1; off < 16; off <<= 1) {
      float oc = __shfl_xor(mnC[r], off);
      int   oi = __shfl_xor(mnI[r], off);
      float oio = __shfl_xor(mnIou[r], off);
      if (oc < mnC[r] || (oc == mnC[r] && oi < mnI[r])) { mnC[r] = oc; mnI[r] = oi; mnIou[r] = oio; }
      smIou[r] += __shfl_xor(smIou[r], off);
      cnt[r]   += __shfl_xor(cnt[r], off);
      int om = __shfl_xor(mMG[r], off);
      mMG[r] = (om < mMG[r]) ? om : mMG[r];
    }
  }

  if (lm == 0) {
    float* out_lab = out;
    float* out_w   = out + (size_t)BN_;
    float* out_box = out + (size_t)2 * BN_;
    float* out_met = out + (size_t)6 * BN_;
#pragma unroll
    for (int r = 0; r < 8; ++r) {
      int n = n0 + r + 8 * hi;
      size_t o = (size_t)b * N_ + n;
      bool multi = cnt[r] > 1;
      bool fg; int mg; float met;
      if (multi)            { fg = true;  mg = mnI[r]; met = mnIou[r]; }
      else if (cnt[r] == 1) { fg = true;  mg = mMG[r]; met = smIou[r]; }
      else                  { fg = false; mg = 0;      met = 0.f; }
      if (mg < 0 || mg >= G_) mg = 0;
      out_lab[o] = fg ? (float)gl[mg] : (float)NC_;
      out_w[o] = 1.f;
      out_box[o * 4 + 0] = fg ? gb[mg * 4 + 0] : 0.f;
      out_box[o * 4 + 1] = fg ? gb[mg * 4 + 1] : 0.f;
      out_box[o * 4 + 2] = fg ? gb[mg * 4 + 2] : 0.f;
      out_box[o * 4 + 3] = fg ? gb[mg * 4 + 3] : 0.f;
      out_met[o] = fg ? met : 0.f;
    }
  }
}

// ---------------------------------------------------------------- launch
extern "C" void kernel_launch(void* const* d_in, const int* in_sizes, int n_in,
                              void* d_out, int out_size, void* d_ws, size_t ws_size,
                              hipStream_t stream) {
  const float* pred_b = (const float*)d_in[0];
  const float* pred_s = (const float*)d_in[1];
  const float* priors = (const float*)d_in[2];
  const int*   gt_lab = (const int*)d_in[3];
  const float* gt_b   = (const float*)d_in[4];
  const float* flag   = (const float*)d_in[5];

  float*    vmask = (float*)d_ws;                                 // BN floats
  unsigned* mmask = (unsigned*)((char*)d_ws + (size_t)BN_ * 4);   // BN*4 u32
  float*    out   = (float*)d_out;

  dim3 g0((N_ + 255) / 256, B_);
  k0_validmask<<<g0, 256, 0, stream>>>(priors, gt_b, flag, vmask, mmask);
  k1_column<<<B_ * G_, 256, 0, stream>>>(pred_b, pred_s, priors, gt_lab, gt_b,
                                         flag, vmask, mmask);
  k2_assign<<<B_ * NT_, 32, 0, stream>>>(pred_b, pred_s, priors, gt_lab, gt_b,
                                         vmask, mmask, out);
}
